// SHAAttention_3762391351923
// MI455X (gfx1250) — matrix operand order repaired and checked by reading, not compiled, not run on hardware
//
#include <hip/hip_runtime.h>
#include <hip/hip_bf16.h>
#include <stdint.h>

#define HEADS 8
#define DHEAD 64
#define CCH   512
#define BB    2
#define HWSZ  2304            // 48*48
#define QSCALE 0.125f         // 1/sqrt(64)

typedef __attribute__((ext_vector_type(16))) __bf16 bf16x16;
typedef __attribute__((ext_vector_type(8)))  float  f32x8;

union FragAB {
    bf16x16 v;
    uint4   q[2];
    unsigned short h[16];
};

__device__ __forceinline__ unsigned short f2bf(float f) {
    union { float f; unsigned int u; } c; c.f = f;
    unsigned int u = c.u;
    u += 0x7fffu + ((u >> 16) & 1u);      // round-to-nearest-even
    return (unsigned short)(u >> 16);
}

__device__ __forceinline__ f32x8 wmma_bf16(const FragAB& a, const FragAB& b, f32x8 c) {
    return __builtin_amdgcn_wmma_f32_16x16x32_bf16(
        false, a.v, false, b.v, (short)0, c, false, false);
}

// ---------------- prep: f32 -> bf16 conversions ----------------

__global__ void cvt_bf16_kernel(const float* __restrict__ src,
                                unsigned short* __restrict__ dst, int n) {
    int i = blockIdx.x * blockDim.x + threadIdx.x;
    if (i < n) dst[i] = f2bf(src[i]);
}

// x [B][C][HW] f32  ->  xT [B][HW][C] bf16
__global__ void transpose_x_kernel(const float* __restrict__ x,
                                   unsigned short* __restrict__ xT) {
    int i = blockIdx.x * blockDim.x + threadIdx.x;
    if (i >= BB * HWSZ * CCH) return;
    int c = i % CCH;
    int s = (i / CCH) % HWSZ;
    int b = i / (CCH * HWSZ);
    xT[i] = f2bf(x[((size_t)b * CCH + c) * HWSZ + s]);
}

// ---------------- Q / K projection ----------------
// D(M=d, N=s) = Wq[64x512] x xT(as B)   ->  out[b][h][s][64] row-major bf16
__global__ void proj_qk_kernel(const unsigned short* __restrict__ xT,   // [B][HW][C]
                               const unsigned short* __restrict__ Wb,   // [H][64][C]
                               const float* __restrict__ bias,          // [H][64]
                               unsigned short* __restrict__ out,        // [B][H][HW][64]
                               float scale) {
    const int lane = threadIdx.x & 31;
    const int wave = threadIdx.x >> 5;          // d-tile (0..3)
    int id = blockIdx.x;
    const int sblk = id % (HWSZ / 64); id /= (HWSZ / 64);
    const int head = id % HEADS;       id /= HEADS;
    const int b    = id;

    const int l15  = lane & 15;
    const int klo8 = (lane >> 4) * 8;
    const int kb16 = (lane >> 4) * 16;
    const int arow = wave * 16 + l15;

    const unsigned short* A  = Wb + (size_t)head * DHEAD * CCH;
    const unsigned short* Bp = xT + (size_t)b * HWSZ * CCH;

    f32x8 acc[4];
    for (int f = 0; f < 4; ++f) for (int i = 0; i < 8; ++i) acc[f][i] = 0.f;

    for (int c0 = 0; c0 < CCH; c0 += 32) {
        FragAB af;
        af.q[0] = *(const uint4*)(A + (size_t)arow * CCH + c0 + klo8);
        af.q[1] = *(const uint4*)(A + (size_t)arow * CCH + c0 + 16 + klo8);
        #pragma unroll
        for (int f = 0; f < 4; ++f) {
            const int scol = sblk * 64 + f * 16 + l15;
            FragAB bfr;
            bfr.q[0] = *(const uint4*)(Bp + (size_t)scol * CCH + c0 + klo8);
            bfr.q[1] = *(const uint4*)(Bp + (size_t)scol * CCH + c0 + 16 + klo8);
            acc[f] = wmma_bf16(af, bfr, acc[f]);
        }
    }

    const int dbase = wave * 16 + klo8;
    unsigned short* O = out + ((size_t)b * HEADS + head) * HWSZ * DHEAD;
    #pragma unroll
    for (int f = 0; f < 4; ++f) {
        const int scol = sblk * 64 + f * 16 + l15;
        union { uint4 q; unsigned short h[8]; } st;
        #pragma unroll
        for (int i = 0; i < 8; ++i)
            st.h[i] = f2bf((acc[f][i] + bias[head * DHEAD + dbase + i]) * scale);
        *(uint4*)(O + (size_t)scol * DHEAD + dbase) = st.q;
    }
}

// ---------------- V projection (stores V transposed [64][HW]) ----------------
__global__ void proj_v_kernel(const unsigned short* __restrict__ xT,   // [B][HW][C]
                              const unsigned short* __restrict__ Wb,   // [H][64][C]
                              const float* __restrict__ bias,          // [H][64]
                              unsigned short* __restrict__ Vt) {       // [B][H][64][HW]
    const int lane = threadIdx.x & 31;
    const int wave = threadIdx.x >> 5;          // s-tile (0..3)
    int id = blockIdx.x;
    const int sblk = id % (HWSZ / 64); id /= (HWSZ / 64);
    const int head = id % HEADS;       id /= HEADS;
    const int b    = id;

    const int l15  = lane & 15;
    const int klo8 = (lane >> 4) * 8;
    const int kb16 = (lane >> 4) * 16;
    const int arow = sblk * 64 + wave * 16 + l15;

    const unsigned short* A  = xT + (size_t)b * HWSZ * CCH;
    const unsigned short* Bp = Wb + (size_t)head * DHEAD * CCH;

    f32x8 acc[4];
    for (int f = 0; f < 4; ++f) for (int i = 0; i < 8; ++i) acc[f][i] = 0.f;

    for (int c0 = 0; c0 < CCH; c0 += 32) {
        FragAB af;
        af.q[0] = *(const uint4*)(A + (size_t)arow * CCH + c0 + klo8);
        af.q[1] = *(const uint4*)(A + (size_t)arow * CCH + c0 + 16 + klo8);
        #pragma unroll
        for (int f = 0; f < 4; ++f) {
            const int dd = f * 16 + l15;
            FragAB bfr;
            bfr.q[0] = *(const uint4*)(Bp + (size_t)dd * CCH + c0 + klo8);
            bfr.q[1] = *(const uint4*)(Bp + (size_t)dd * CCH + c0 + 16 + klo8);
            acc[f] = wmma_bf16(af, bfr, acc[f]);
        }
    }

    const int sbase = sblk * 64 + wave * 16 + klo8;
    unsigned short* O = Vt + ((size_t)b * HEADS + head) * DHEAD * HWSZ;
    #pragma unroll
    for (int f = 0; f < 4; ++f) {
        const int dd = f * 16 + l15;
        union { uint4 q; unsigned short h[8]; } st;
        #pragma unroll
        for (int i = 0; i < 8; ++i)
            st.h[i] = f2bf(acc[f][i] + bias[head * DHEAD + dd]);
        *(uint4*)(O + (size_t)dd * HWSZ + sbase) = st.q;
    }
}

// ---------------- flash attention (fused QK^T / softmax / PV) ----------------
__global__ void attn_kernel(const unsigned short* __restrict__ Q,   // [B][H][HW][64]
                            const unsigned short* __restrict__ K,   // [B][H][HW][64]
                            const unsigned short* __restrict__ Vt,  // [B][H][64][HW]
                            unsigned short* __restrict__ attn) {    // [B][HW][H*64]
    __shared__ unsigned short lds_p[4][16][32];   // per-wave P tile, 4 KB total
    const int lane = threadIdx.x & 31;
    const int wave = threadIdx.x >> 5;
    int id = blockIdx.x;
    const int qblk = id % (HWSZ / 64); id /= (HWSZ / 64);
    const int head = id % HEADS;       id /= HEADS;
    const int b    = id;

    const size_t hb = (size_t)b * HEADS + head;
    const unsigned short* Qp = Q  + hb * HWSZ * DHEAD;
    const unsigned short* Kp = K  + hb * HWSZ * DHEAD;
    const unsigned short* Vp = Vt + hb * DHEAD * HWSZ;

    const int l15  = lane & 15;
    const int klo8 = (lane >> 4) * 8;
    const int kb16 = (lane >> 4) * 16;
    const int qrow = qblk * 64 + wave * 16 + l15;

    FragAB aq[2];
    #pragma unroll
    for (int t = 0; t < 2; ++t) {
        aq[t].q[0] = *(const uint4*)(Qp + (size_t)qrow * DHEAD + t * 32 + klo8);
        aq[t].q[1] = *(const uint4*)(Qp + (size_t)qrow * DHEAD + t * 32 + 16 + klo8);
    }

    float mrow[8], lrow[8];
    #pragma unroll
    for (int i = 0; i < 8; ++i) { mrow[i] = -1e30f; lrow[i] = 0.f; }
    f32x8 o[4];
    for (int f = 0; f < 4; ++f) for (int i = 0; i < 8; ++i) o[f][i] = 0.f;

    for (int sk0 = 0; sk0 < HWSZ; sk0 += 32) {
        f32x8 s[2];
        for (int j = 0; j < 2; ++j) for (int i = 0; i < 8; ++i) s[j][i] = 0.f;
        #pragma unroll
        for (int j = 0; j < 2; ++j) {
            const int skcol = sk0 + j * 16 + l15;
            #pragma unroll
            for (int t = 0; t < 2; ++t) {
                FragAB bk;
                bk.q[0] = *(const uint4*)(Kp + (size_t)skcol * DHEAD + t * 32 + klo8);
                bk.q[1] = *(const uint4*)(Kp + (size_t)skcol * DHEAD + t * 32 + 16 + klo8);
                s[j] = wmma_bf16(aq[t], bk, s[j]);
            }
        }
        // online softmax: row = (lane>=16 ? 8 : 0) + i, cols across 16 lanes
        #pragma unroll
        for (int i = 0; i < 8; ++i) {
            float mx = fmaxf(s[0][i], s[1][i]);
            #pragma unroll
            for (int m = 8; m >= 1; m >>= 1) mx = fmaxf(mx, __shfl_xor(mx, m, 32));
            float mnew  = fmaxf(mrow[i], mx);
            float alpha = __expf(mrow[i] - mnew);
            mrow[i] = mnew;
            float p0 = __expf(s[0][i] - mnew);
            float p1 = __expf(s[1][i] - mnew);
            s[0][i] = p0; s[1][i] = p1;
            float rs = p0 + p1;
            #pragma unroll
            for (int m = 8; m >= 1; m >>= 1) rs += __shfl_xor(rs, m, 32);
            lrow[i] = lrow[i] * alpha + rs;
            #pragma unroll
            for (int f = 0; f < 4; ++f) o[f][i] *= alpha;
        }
        // P (C/D layout) -> LDS row-major -> reload as A-layout fragment
        asm volatile("" ::: "memory");
        #pragma unroll
        for (int j = 0; j < 2; ++j)
            #pragma unroll
            for (int i = 0; i < 8; ++i)
                lds_p[wave][klo8 + i][j * 16 + l15] = f2bf(s[j][i]);
        asm volatile("s_wait_dscnt 0x0" ::: "memory");   // CDNA5 split DS counter
        FragAB ap;
        ap.q[0] = *(const uint4*)(&lds_p[wave][l15][klo8]);
        ap.q[1] = *(const uint4*)(&lds_p[wave][l15][16 + klo8]);
        #pragma unroll
        for (int f = 0; f < 4; ++f) {
            const int dd = f * 16 + l15;
            FragAB bv;
            bv.q[0] = *(const uint4*)(Vp + (size_t)dd * HWSZ + sk0 + klo8);
            bv.q[1] = *(const uint4*)(Vp + (size_t)dd * HWSZ + sk0 + 16 + klo8);
            o[f] = wmma_bf16(ap, bv, o[f]);
        }
    }

    // finalize: O /= l, store into concat-heads buffer (bf16)
    const int rbase = qblk * 64 + wave * 16 + klo8;
    unsigned short* Ap = attn + (size_t)b * HWSZ * (HEADS * DHEAD);
    #pragma unroll
    for (int i = 0; i < 8; ++i) {
        float inv = 1.0f / lrow[i];
        #pragma unroll
        for (int f = 0; f < 4; ++f) {
            const int dd = f * 16 + l15;
            Ap[(size_t)(rbase + i) * (HEADS * DHEAD) + head * DHEAD + dd] =
                f2bf(o[f][i] * inv);
        }
    }
}

// ---------------- output projection: out[b][co][s] = attn @ Wo^T + bo ----------------
__global__ void out_proj_kernel(const unsigned short* __restrict__ attn, // [B][HW][512]
                                const unsigned short* __restrict__ WoB,  // [512][512]
                                const float* __restrict__ bo,            // [512]
                                float* __restrict__ out) {               // [B][C][HW]
    const int lane = threadIdx.x & 31;
    const int wave = threadIdx.x >> 5;
    int id = blockIdx.x;
    const int cblk = id % (CCH / 128);  id /= (CCH / 128);
    const int sblk = id % (HWSZ / 64);  id /= (HWSZ / 64);
    const int b    = id;

    const int l15  = lane & 15;
    const int klo8 = (lane >> 4) * 8;
    const int kb16 = (lane >> 4) * 16;
    const int arow = sblk * 64 + wave * 16 + l15;
    const unsigned short* A = attn + (size_t)b * HWSZ * CCH;

    f32x8 acc[8];
    for (int f = 0; f < 8; ++f) for (int i = 0; i < 8; ++i) acc[f][i] = 0.f;

    for (int j0 = 0; j0 < CCH; j0 += 32) {
        FragAB af;
        af.q[0] = *(const uint4*)(A + (size_t)arow * CCH + j0 + klo8);
        af.q[1] = *(const uint4*)(A + (size_t)arow * CCH + j0 + 16 + klo8);
        #pragma unroll
        for (int f = 0; f < 8; ++f) {
            const int co = cblk * 128 + f * 16 + l15;
            FragAB bfr;
            bfr.q[0] = *(const uint4*)(WoB + (size_t)co * CCH + j0 + klo8);
            bfr.q[1] = *(const uint4*)(WoB + (size_t)co * CCH + j0 + 16 + klo8);
            acc[f] = wmma_bf16(af, bfr, acc[f]);
        }
    }

    const int sbase = sblk * 64 + wave * 16 + klo8;
    #pragma unroll
    for (int f = 0; f < 8; ++f) {
        const int co = cblk * 128 + f * 16 + l15;
        const float bb = bo[co];
        float4 lo = make_float4(acc[f][0] + bb, acc[f][1] + bb, acc[f][2] + bb, acc[f][3] + bb);
        float4 hi = make_float4(acc[f][4] + bb, acc[f][5] + bb, acc[f][6] + bb, acc[f][7] + bb);
        float* dst = out + ((size_t)b * CCH + co) * HWSZ + sbase;
        *(float4*)(dst)     = lo;
        *(float4*)(dst + 4) = hi;
    }
}

// ---------------- host launcher ----------------
extern "C" void kernel_launch(void* const* d_in, const int* in_sizes, int n_in,
                              void* d_out, int out_size, void* d_ws, size_t ws_size,
                              hipStream_t stream) {
    (void)in_sizes; (void)n_in; (void)out_size; (void)ws_size;

    const float* x  = (const float*)d_in[0];
    const float* Wq = (const float*)d_in[1];
    const float* bq = (const float*)d_in[2];
    const float* Wk = (const float*)d_in[3];
    const float* bk = (const float*)d_in[4];
    const float* Wv = (const float*)d_in[5];
    const float* bv = (const float*)d_in[6];
    const float* Wo = (const float*)d_in[7];
    const float* bo = (const float*)d_in[8];

    uint8_t* p = (uint8_t*)d_ws;                       // ~25.7 MB used
    unsigned short* xT   = (unsigned short*)p; p += (size_t)BB * HWSZ * CCH * 2;
    unsigned short* Wqb  = (unsigned short*)p; p += (size_t)HEADS * DHEAD * CCH * 2;
    unsigned short* Wkb  = (unsigned short*)p; p += (size_t)HEADS * DHEAD * CCH * 2;
    unsigned short* Wvb  = (unsigned short*)p; p += (size_t)HEADS * DHEAD * CCH * 2;
    unsigned short* Wob  = (unsigned short*)p; p += (size_t)CCH * CCH * 2;
    unsigned short* Qb   = (unsigned short*)p; p += (size_t)BB * HEADS * HWSZ * DHEAD * 2;
    unsigned short* Kb   = (unsigned short*)p; p += (size_t)BB * HEADS * HWSZ * DHEAD * 2;
    unsigned short* Vtb  = (unsigned short*)p; p += (size_t)BB * HEADS * DHEAD * HWSZ * 2;
    unsigned short* Atb  = (unsigned short*)p; p += (size_t)BB * HWSZ * CCH * 2;

    const int nW = HEADS * DHEAD * CCH;     // 262144
    const int nX = BB * HWSZ * CCH;         // 2359296

    transpose_x_kernel<<<(nX + 255) / 256, 256, 0, stream>>>(x, xT);
    cvt_bf16_kernel<<<(nW + 255) / 256, 256, 0, stream>>>(Wq, Wqb, nW);
    cvt_bf16_kernel<<<(nW + 255) / 256, 256, 0, stream>>>(Wk, Wkb, nW);
    cvt_bf16_kernel<<<(nW + 255) / 256, 256, 0, stream>>>(Wv, Wvb, nW);
    cvt_bf16_kernel<<<(CCH * CCH + 255) / 256, 256, 0, stream>>>(Wo, Wob, CCH * CCH);

    const int gProj = BB * HEADS * (HWSZ / 64);         // 576 blocks
    proj_qk_kernel<<<gProj, 128, 0, stream>>>(xT, Wqb, bq, Qb, QSCALE);
    proj_qk_kernel<<<gProj, 128, 0, stream>>>(xT, Wkb, bk, Kb, 1.0f);
    proj_v_kernel <<<gProj, 128, 0, stream>>>(xT, Wvb, bv, Vtb);

    attn_kernel<<<gProj, 128, 0, stream>>>(Qb, Kb, Vtb, Atb);

    const int gOut = BB * (HWSZ / 64) * (CCH / 128);    // 288 blocks
    out_proj_kernel<<<gOut, 128, 0, stream>>>(Atb, Wob, bo, (float*)d_out);
}
